// GATClassifier_64613488001235
// MI455X (gfx1250) — matrix operand order repaired and checked by reading, not compiled, not run on hardware
//
#include <hip/hip_runtime.h>
#include <hip/hip_bf16.h>

typedef _Float16 f16;
typedef __attribute__((ext_vector_type(16))) _Float16 v16h;
typedef __attribute__((ext_vector_type(8)))  float    v8f;

#define NN 20000
#define EE 320000
#define ET (EE + NN)
#define HID 128
#define ZW  512          // HEADS*HID
#define NH  4
#define NL  3
#define NG  64
#define NEG_SLOPE 0.2f
#define LN_EPS 1e-5f

// ---------------- device helpers ----------------

__device__ inline float wave_sum(float v) {
#pragma unroll
  for (int off = 16; off > 0; off >>= 1) v += __shfl_xor(v, off, 32);
  return v;
}

__device__ inline void atomicMaxF(float* a, float v) {
  if (v >= 0.f) atomicMax((int*)a, __float_as_int(v));
  else          atomicMin((unsigned int*)a, __float_as_uint(v));
}

// ---------------- prep kernels ----------------

__global__ void prep_edges_kernel(const int* __restrict__ eidx, int* __restrict__ src,
                                  int* __restrict__ dst) {
  int i = blockIdx.x * blockDim.x + threadIdx.x;
  if (i < EE) { src[i] = eidx[i]; dst[i] = eidx[EE + i]; }
  else if (i < ET) { int n = i - EE; src[i] = n; dst[i] = n; }
}

__global__ void convert_f16_kernel(const float* __restrict__ in, f16* __restrict__ out, int n) {
  int i = blockIdx.x * blockDim.x + threadIdx.x;
  if (i < n) out[i] = (f16)in[i];
}

// W is [K,Nc] row-major fp32 -> Wt is [Nc,K] row-major f16
__global__ void transpose_convert_kernel(const float* __restrict__ W, f16* __restrict__ Wt,
                                         int K, int Nc) {
  int i = blockIdx.x * blockDim.x + threadIdx.x;
  if (i >= K * Nc) return;
  int k = i / Nc, n = i % Nc;
  Wt[(size_t)n * K + k] = (f16)W[i];
}

__global__ void fill_kernel(float* __restrict__ p, float v, int n) {
  int i = blockIdx.x * blockDim.x + threadIdx.x;
  if (i < n) p[i] = v;
}

__global__ void bias_broadcast_kernel(float* __restrict__ agg, const float* __restrict__ bias) {
  int i = blockIdx.x * blockDim.x + threadIdx.x;
  if (i < NN * HID) agg[i] = bias[i & (HID - 1)];
}

// ---------------- WMMA GEMM: C[M,Nc] = A[M,K](f16) * Bt[Nc,K]^T(f16) (+bias)(relu) ----------------
// One 16x16 tile per wave, K stepped by 32.

__global__ void gemm_wmma_kernel(const f16* __restrict__ A, const f16* __restrict__ Bt,
                                 const float* __restrict__ bias,
                                 float* __restrict__ Cf32, f16* __restrict__ Cf16,
                                 int M, int K, int Nc, int relu) {
  int wave = threadIdx.x >> 5;
  int lane = threadIdx.x & 31;
  int nT = Nc >> 4;
  int mT = M >> 4;
  int tile = blockIdx.x * (blockDim.x >> 5) + wave;
  if (tile >= mT * nT) return;              // wave-uniform: EXEC stays all-ones
  int mt = tile / nT, nt = tile % nT;
  int l16  = lane & 15;
  int hsel = lane >> 4;                     // 0 or 1

  const float4* arow = (const float4*)(A  + (size_t)((mt << 4) + l16) * K);
  const float4* brow = (const float4*)(Bt + (size_t)((nt << 4) + l16) * K);

  v8f c = {};
  for (int kk = 0; kk < K; kk += 32) {
    int cb = kk >> 3;                       // 8-half (16B) chunk index
    union { v16h v; float4 f[2]; } a, b;
    // A 16-bit layout: lanes 0-15 hold K {kk..kk+7, kk+16..kk+23}; lanes 16-31 the other halves
    a.f[0] = arow[cb + hsel];
    a.f[1] = arow[cb + 2 + hsel];
    // B layout: same K order as A: lanes 0-15 hold K {kk..kk+7, kk+16..kk+23}; lanes 16-31 the other halves
    b.f[0] = brow[cb + hsel];
    b.f[1] = brow[cb + 2 + hsel];
    c = __builtin_amdgcn_wmma_f32_16x16x32_f16(false, a.v, false, b.v, (short)0, c,
                                               false, false);
  }

  int n = (nt << 4) + l16;
  float bv = bias ? bias[n] : 0.f;
  int rowBase = (mt << 4) + (hsel << 3);
#pragma unroll
  for (int r = 0; r < 8; r++) {
    float y = c[r] + bv;
    if (relu) y = fmaxf(y, 0.f);
    size_t idx = (size_t)(rowBase + r) * Nc + n;
    if (Cf32) Cf32[idx] = y;
    if (Cf16) Cf16[idx] = (f16)y;
  }
}

// ---------------- attention ----------------

// a_src[n,h] = sum_c z[n,h,c]*att_src[h,c] ; same for dst. One wave per (n,h).
__global__ void att_score_kernel(const f16* __restrict__ z, const float* __restrict__ as_w,
                                 const float* __restrict__ ad_w,
                                 float* __restrict__ a_src, float* __restrict__ a_dst) {
  int wid  = (blockIdx.x * blockDim.x + threadIdx.x) >> 5;
  int lane = threadIdx.x & 31;
  if (wid >= NN * NH) return;
  int n = wid >> 2, h = wid & 3;
  const f16* zr = z + (size_t)n * ZW + h * HID;
  const float* as = as_w + h * HID;
  const float* ad = ad_w + h * HID;
  float ss = 0.f, sd = 0.f;
#pragma unroll
  for (int j = 0; j < 4; j++) {
    int c = lane + 32 * j;
    float zv = (float)zr[c];
    ss += zv * as[c];
    sd += zv * ad[c];
  }
  ss = wave_sum(ss); sd = wave_sum(sd);
  if (lane == 0) { a_src[wid] = ss; a_dst[wid] = sd; }
}

__device__ inline float lrelu(float v) { return v > 0.f ? v : NEG_SLOPE * v; }

__global__ void edge_max_kernel(const int* __restrict__ src, const int* __restrict__ dst,
                                const float* __restrict__ a_src, const float* __restrict__ a_dst,
                                float* __restrict__ e_max) {
  int i = blockIdx.x * blockDim.x + threadIdx.x;
  if (i >= ET * NH) return;
  int e = i >> 2, h = i & 3;
  int s = src[e], d = dst[e];
  float v = lrelu(a_src[s * NH + h] + a_dst[d * NH + h]);
  atomicMaxF(&e_max[d * NH + h], v);
}

__global__ void edge_exp_kernel(const int* __restrict__ src, const int* __restrict__ dst,
                                const float* __restrict__ a_src, const float* __restrict__ a_dst,
                                const float* __restrict__ e_max,
                                float* __restrict__ ex, float* __restrict__ denom) {
  int i = blockIdx.x * blockDim.x + threadIdx.x;
  if (i >= ET * NH) return;
  int e = i >> 2, h = i & 3;
  int s = src[e], d = dst[e];
  float v = lrelu(a_src[s * NH + h] + a_dst[d * NH + h]);
  float x = __expf(v - e_max[d * NH + h]);
  ex[i] = x;
  atomicAdd(&denom[d * NH + h], x);
}

// One wave per edge; fuse alpha normalization + head mean into the scatter.
__global__ void edge_agg_kernel(const int* __restrict__ src, const int* __restrict__ dst,
                                const f16* __restrict__ z, const float* __restrict__ ex,
                                const float* __restrict__ denom, float* __restrict__ agg) {
  int wid  = (blockIdx.x * blockDim.x + threadIdx.x) >> 5;
  int lane = threadIdx.x & 31;
  if (wid >= ET) return;
  int s = src[wid], d = dst[wid];
  float al[NH];
#pragma unroll
  for (int h = 0; h < NH; h++) al[h] = ex[wid * NH + h] / denom[d * NH + h];
  const f16* zr = z + (size_t)s * ZW;
#pragma unroll
  for (int j = 0; j < 4; j++) {
    int c = lane + 32 * j;
    float acc = 0.f;
#pragma unroll
    for (int h = 0; h < NH; h++) acc += al[h] * (float)zr[h * HID + c];
    atomicAdd(&agg[(size_t)d * HID + c], 0.25f * acc);  // 0.25 = head mean
  }
}

// ---------------- layernorm (+relu, +optional residual), one wave per row ----------------

__global__ void ln_row_kernel(const float* __restrict__ in, const float* __restrict__ gamma,
                              const float* __restrict__ beta, const float* __restrict__ residual,
                              float* __restrict__ out_f32, f16* __restrict__ out_f16, int nrows) {
  int wid  = (blockIdx.x * blockDim.x + threadIdx.x) >> 5;
  int lane = threadIdx.x & 31;
  if (wid >= nrows) return;
  const float* r = in + (size_t)wid * HID;
  float v[4]; float s = 0.f;
#pragma unroll
  for (int j = 0; j < 4; j++) { v[j] = r[lane + 32 * j]; s += v[j]; }
  s = wave_sum(s);
  float mu = s * (1.f / HID);
  float q = 0.f;
#pragma unroll
  for (int j = 0; j < 4; j++) { float d = v[j] - mu; q += d * d; }
  q = wave_sum(q);
  float inv = rsqrtf(q * (1.f / HID) + LN_EPS);
#pragma unroll
  for (int j = 0; j < 4; j++) {
    int c = lane + 32 * j;
    float y = (v[j] - mu) * inv * gamma[c] + beta[c];
    y = fmaxf(y, 0.f);
    if (residual) y += residual[(size_t)wid * HID + c];
    if (out_f32) out_f32[(size_t)wid * HID + c] = y;
    if (out_f16) out_f16[(size_t)wid * HID + c] = (f16)y;
  }
}

// ---------------- pooling ----------------

__global__ void pool_kernel(const int* __restrict__ batch, const float* __restrict__ h,
                            float* __restrict__ gsum, float* __restrict__ gmax,
                            float* __restrict__ cnt) {
  int wid  = (blockIdx.x * blockDim.x + threadIdx.x) >> 5;
  int lane = threadIdx.x & 31;
  if (wid >= NN) return;
  int b = batch[wid];
  const float* r = h + (size_t)wid * HID;
#pragma unroll
  for (int j = 0; j < 4; j++) {
    int c = lane + 32 * j;
    float v = r[c];
    atomicAdd(&gsum[b * HID + c], v);
    atomicMaxF(&gmax[b * HID + c], v);
  }
  if (lane == 0) atomicAdd(&cnt[b], 1.0f);
}

__global__ void gemb_kernel(const float* __restrict__ gsum, const float* __restrict__ gmax,
                            const float* __restrict__ cnt, float* __restrict__ gemb) {
  int i = blockIdx.x * blockDim.x + threadIdx.x;
  if (i >= NG * HID) return;
  int g = i >> 7, c = i & (HID - 1);
  gemb[g * 2 * HID + c]       = gsum[i] / cnt[g];
  gemb[g * 2 * HID + HID + c] = gmax[i];
}

// ---------------- graph head: one block (128 threads) per graph ----------------

__global__ void graph_head_kernel(const float* __restrict__ gemb,
                                  const float* __restrict__ w1, const float* __restrict__ b1,
                                  const float* __restrict__ lng, const float* __restrict__ lnb,
                                  const float* __restrict__ w2, const float* __restrict__ b2,
                                  const float* __restrict__ w3, const float* __restrict__ b3,
                                  float* __restrict__ out) {
  __shared__ float emb[2 * HID];
  __shared__ float s1[HID];
  __shared__ float red[HID];
  int g = blockIdx.x, t = threadIdx.x;
  emb[t] = gemb[g * 2 * HID + t];
  emb[t + HID] = gemb[g * 2 * HID + HID + t];
  __syncthreads();
  float acc = b1[t];
  for (int k = 0; k < 2 * HID; k++) acc += emb[k] * w1[k * HID + t];
  red[t] = acc; __syncthreads();
  for (int s = 64; s > 0; s >>= 1) { if (t < s) red[t] += red[t + s]; __syncthreads(); }
  float mu = red[0] / (float)HID; __syncthreads();
  float d = acc - mu;
  red[t] = d * d; __syncthreads();
  for (int s = 64; s > 0; s >>= 1) { if (t < s) red[t] += red[t + s]; __syncthreads(); }
  float var = red[0] / (float)HID;
  float y = d * rsqrtf(var + LN_EPS) * lng[t] + lnb[t];
  s1[t] = fmaxf(y, 0.f);
  __syncthreads();
  if (t < 64) {
    float a2 = b2[t];
    for (int k = 0; k < HID; k++) a2 += s1[k] * w2[k * 64 + t];
    red[t] = fmaxf(a2, 0.f);
  }
  __syncthreads();
  if (t < 2) {
    float a3 = b3[t];
    for (int k = 0; k < 64; k++) a3 += red[k] * w3[k * 2 + t];
    out[g * 2 + t] = a3;
  }
}

// ---------------- node head final 64->2 ----------------

__global__ void nh3_kernel(const float* __restrict__ t2, const float* __restrict__ w3,
                           const float* __restrict__ b3, float* __restrict__ out) {
  int n = blockIdx.x * blockDim.x + threadIdx.x;
  if (n >= NN) return;
  float a0 = b3[0], a1 = b3[1];
  const float* r = t2 + (size_t)n * 64;
  for (int k = 0; k < 64; k++) { float v = r[k]; a0 += v * w3[k * 2]; a1 += v * w3[k * 2 + 1]; }
  out[n * 2]     = a0;
  out[n * 2 + 1] = a1;
}

// ---------------- host ----------------

static inline void launch_gemm(const f16* A, const f16* Bt, const float* bias,
                               float* Cf, f16* Ch, int M, int K, int Nc, int relu,
                               hipStream_t s) {
  int tiles = (M / 16) * (Nc / 16);
  int blocks = (tiles + 7) / 8;
  gemm_wmma_kernel<<<blocks, 256, 0, s>>>(A, Bt, bias, Cf, Ch, M, K, Nc, relu);
}

extern "C" void kernel_launch(void* const* d_in, const int* in_sizes, int n_in,
                              void* d_out, int out_size, void* d_ws, size_t ws_size,
                              hipStream_t stream) {
  (void)in_sizes; (void)n_in; (void)out_size; (void)ws_size;
  const float* x       = (const float*)d_in[0];
  const int*   eidx    = (const int*)d_in[1];
  const int*   batch   = (const int*)d_in[2];
  const float* proj_w  = (const float*)d_in[3];
  const float* proj_b  = (const float*)d_in[4];
  const float* gat_w   = (const float*)d_in[5];
  const float* att_s   = (const float*)d_in[6];
  const float* att_d   = (const float*)d_in[7];
  const float* gat_b   = (const float*)d_in[8];
  const float* ln_g    = (const float*)d_in[9];
  const float* ln_b    = (const float*)d_in[10];
  const float* gh_w1   = (const float*)d_in[11];
  const float* gh_b1   = (const float*)d_in[12];
  const float* gh_lng  = (const float*)d_in[13];
  const float* gh_lnb  = (const float*)d_in[14];
  const float* gh_w2   = (const float*)d_in[15];
  const float* gh_b2   = (const float*)d_in[16];
  const float* gh_w3   = (const float*)d_in[17];
  const float* gh_b3   = (const float*)d_in[18];
  const float* nh_w1   = (const float*)d_in[19];
  const float* nh_b1   = (const float*)d_in[20];
  const float* nh_lng  = (const float*)d_in[21];
  const float* nh_lnb  = (const float*)d_in[22];
  const float* nh_w2   = (const float*)d_in[23];
  const float* nh_b2   = (const float*)d_in[24];
  const float* nh_w3   = (const float*)d_in[25];
  const float* nh_b3   = (const float*)d_in[26];
  float* out = (float*)d_out;

  // ---- workspace bump allocator (256B aligned) ----
  char* base = (char*)d_ws; size_t off = 0;
  auto alloc = [&](size_t bytes) -> void* {
    void* p = base + off; off = (off + bytes + 255) & ~(size_t)255; return p;
  };
  int*   srcI  = (int*)  alloc((size_t)ET * 4);
  int*   dstI  = (int*)  alloc((size_t)ET * 4);
  f16*   x16   = (f16*)  alloc((size_t)NN * HID * 2);
  float* h     = (float*)alloc((size_t)NN * HID * 4);
  f16*   h16   = (f16*)  alloc((size_t)NN * HID * 2);
  f16*   z16   = (f16*)  alloc((size_t)NN * ZW * 2);   // reused later as t2 [NN,64] f32
  float* agg   = (float*)alloc((size_t)NN * HID * 4);
  float* aS    = (float*)alloc((size_t)NN * NH * 4);
  float* aD    = (float*)alloc((size_t)NN * NH * 4);
  float* eMax  = (float*)alloc((size_t)NN * NH * 4);
  float* den   = (float*)alloc((size_t)NN * NH * 4);
  float* ex    = (float*)alloc((size_t)ET * NH * 4);
  f16*   projT = (f16*)  alloc((size_t)HID * HID * 2);
  f16*   gatT  = (f16*)  alloc((size_t)NL * ZW * HID * 2);
  f16*   nh1T  = (f16*)  alloc((size_t)HID * HID * 2);
  f16*   nh2T  = (f16*)  alloc((size_t)64 * HID * 2);
  float* gsum  = (float*)alloc((size_t)NG * HID * 4);
  float* gmax  = (float*)alloc((size_t)NG * HID * 4);
  float* cnt   = (float*)alloc((size_t)NG * 4);
  float* gemb  = (float*)alloc((size_t)NG * 2 * HID * 4);

  const int B = 256;
  auto gr = [](long long n, int b) { return (unsigned)((n + b - 1) / b); };

  // ---- prep ----
  prep_edges_kernel<<<gr(ET, B), B, 0, stream>>>(eidx, srcI, dstI);
  convert_f16_kernel<<<gr((long long)NN * HID, B), B, 0, stream>>>(x, x16, NN * HID);
  transpose_convert_kernel<<<gr(HID * HID, B), B, 0, stream>>>(proj_w, projT, HID, HID);
  for (int l = 0; l < NL; l++)
    transpose_convert_kernel<<<gr(HID * ZW, B), B, 0, stream>>>(
        gat_w + (size_t)l * HID * ZW, gatT + (size_t)l * ZW * HID, HID, ZW);
  transpose_convert_kernel<<<gr(HID * HID, B), B, 0, stream>>>(nh_w1, nh1T, HID, HID);
  transpose_convert_kernel<<<gr(HID * 64, B), B, 0, stream>>>(nh_w2, nh2T, HID, 64);

  // ---- projection: h = relu(x @ proj_w + b) ----
  launch_gemm(x16, projT, proj_b, h, h16, NN, HID, HID, 1, stream);

  // ---- GAT layers ----
  for (int l = 0; l < NL; l++) {
    // z = h @ W  -> f16 [N, 4*128]
    launch_gemm(h16, gatT + (size_t)l * ZW * HID, nullptr, nullptr, z16, NN, HID, ZW, 0, stream);
    att_score_kernel<<<gr((long long)NN * NH, 8), B, 0, stream>>>(
        z16, att_s + (size_t)l * NH * HID, att_d + (size_t)l * NH * HID, aS, aD);
    fill_kernel<<<gr(NN * NH, B), B, 0, stream>>>(eMax, -3.4e38f, NN * NH);
    fill_kernel<<<gr(NN * NH, B), B, 0, stream>>>(den, 0.f, NN * NH);
    bias_broadcast_kernel<<<gr((long long)NN * HID, B), B, 0, stream>>>(agg, gat_b + (size_t)l * HID);
    edge_max_kernel<<<gr((long long)ET * NH, B), B, 0, stream>>>(srcI, dstI, aS, aD, eMax);
    edge_exp_kernel<<<gr((long long)ET * NH, B), B, 0, stream>>>(srcI, dstI, aS, aD, eMax, ex, den);
    edge_agg_kernel<<<gr(ET, 8), B, 0, stream>>>(srcI, dstI, z16, ex, den, agg);
    // h = relu(LN(agg)) + identity ; refresh h16
    ln_row_kernel<<<gr(NN, 8), B, 0, stream>>>(
        agg, ln_g + (size_t)l * HID, ln_b + (size_t)l * HID, h, h, h16, NN);
  }

  // ---- pooling ----
  fill_kernel<<<gr(NG * HID, B), B, 0, stream>>>(gsum, 0.f, NG * HID);
  fill_kernel<<<gr(NG * HID, B), B, 0, stream>>>(gmax, -3.4e38f, NG * HID);
  fill_kernel<<<gr(NG, B), B, 0, stream>>>(cnt, 0.f, NG);
  pool_kernel<<<gr(NN, 8), B, 0, stream>>>(batch, h, gsum, gmax, cnt);
  gemb_kernel<<<gr(NG * HID, B), B, 0, stream>>>(gsum, gmax, cnt, gemb);

  // ---- graph head -> out[0:128] ----
  graph_head_kernel<<<NG, HID, 0, stream>>>(gemb, gh_w1, gh_b1, gh_lng, gh_lnb,
                                            gh_w2, gh_b2, gh_w3, gh_b3, out);

  // ---- node head -> out[128:...] ----
  launch_gemm(h16, nh1T, nh_b1, agg, nullptr, NN, HID, HID, 0, stream);          // t1 = h@W1+b1
  ln_row_kernel<<<gr(NN, 8), B, 0, stream>>>(agg, nh_lng, nh_lnb, nullptr,
                                             nullptr, h16, NN);                  // relu(LN(t1)) -> f16
  float* t2 = (float*)z16;                                                       // reuse z space
  launch_gemm(h16, nh2T, nh_b2, t2, nullptr, NN, HID, 64, 1, stream);            // relu(t@W2+b2)
  nh3_kernel<<<gr(NN, B), B, 0, stream>>>(t2, nh_w3, nh_b3, out + 2 * NG);
}
